// SRGNN_12987981103276
// MI455X (gfx1250) — matrix operand order repaired and checked by reading, not compiled, not run on hardware
//
#include <hip/hip_runtime.h>
#include <hip/hip_bf16.h>
#include <math.h>

#define B_  512
#define N_  50
#define L_  50
#define H_  128
#define NN_ 100000

typedef __attribute__((ext_vector_type(16))) _Float16 v16h;
typedef __attribute__((ext_vector_type(8)))  _Float16 v8h;
typedef __attribute__((ext_vector_type(8)))  float    v8f;

// ---------------- WMMA fragment loaders (ISA 7.12.2 layouts) ----------------
// A-matrix 16x32 f16: lane<16 (M=lane): halves 0..7 = K k0+0..7, halves 8..15 = K k0+16..23
//                     lane>=16 (M=lane-16): halves 0..7 = K k0+8..15, halves 8..15 = K k0+24..31
__device__ __forceinline__ v16h load_fragA(const _Float16* rowbase, int k0, int lane) {
  const _Float16* q = rowbase + k0 + ((lane & 16) ? 8 : 0);
  v8h lo = *(const v8h*)(q);
  v8h hi = *(const v8h*)(q + 16);
  v16h r;
#pragma unroll
  for (int i = 0; i < 8; ++i) { r[i] = lo[i]; r[i + 8] = hi[i]; }
  return r;
}

// B-matrix 32x16 f16 (column n = row of Bt): lane<16 (N=lane): halves 0..7 = K k0+0..7, halves 8..15 = K k0+16..23
//                                            lane>=16: halves 0..7 = K k0+8..15, halves 8..15 = K k0+24..31
__device__ __forceinline__ v16h load_fragB(const _Float16* rowbase, int k0, int lane) {
  const _Float16* q = rowbase + k0 + ((lane & 16) ? 8 : 0);
  v8h lo = *(const v8h*)(q);
  v8h hi = *(const v8h*)(q + 16);
  v16h r;
#pragma unroll
  for (int i = 0; i < 8; ++i) { r[i] = lo[i]; r[i + 8] = hi[i]; }
  return r;
}

// ---------------- generic WMMA GEMM: C(MxN) = A(MxK) @ Bt(NxK)^T + bias -----
// A, Bt row-major f16; C row-major f32. One 16x16 tile per wave; K compile-time,
// multiple of 32, fully unrolled. N-edge handled by CLAMPING the B row index
// (WMMA columns are independent, so a clamped/duplicate column only produces
// values in output columns that the store guard discards) -> no EXEC
// manipulation, no zero-fills, no branches in the WMMA sequence.
template <int K>
__global__ void wmma_gemm_bias_t(const _Float16* __restrict__ A,
                                 const _Float16* __restrict__ Bt,
                                 const float* __restrict__ bias,
                                 float* __restrict__ C,
                                 int M, int N, int tilesN, int totalTiles) {
  int tile = blockIdx.x * blockDim.y + threadIdx.y;   // wave-uniform
  if (tile >= totalTiles) return;
  int lane = threadIdx.x & 31;
  int tM = tile / tilesN;
  int tN = tile - tM * tilesN;

  int r = lane & 15;
  const _Float16* arow = A + (size_t)(tM * 16 + r) * K;
  int bn = tN * 16 + r;
  bn = (bn < N) ? bn : (N - 1);                 // clamp: branchless edge handling
  const _Float16* brow = Bt + (size_t)bn * K;

  v8f acc = {};
#pragma unroll
  for (int k0 = 0; k0 < K; k0 += 32) {
    v16h a = load_fragA(arow, k0, lane);
    v16h b = load_fragB(brow, k0, lane);
    acc = __builtin_amdgcn_wmma_f32_16x16x32_f16(
        false, a, false, b, (short)0, acc, false, false);
  }

  // C/D layout: lane<16 -> (M=tM*16+j, N=lane); lane>=16 -> (M=tM*16+8+j, N=lane-16)
  int mbase = tM * 16 + ((lane & 16) ? 8 : 0);
  int nn = tN * 16 + (lane & 15);
  if (nn < N) {
    float bv = bias ? bias[nn] : 0.0f;
#pragma unroll
    for (int j = 0; j < 8; ++j)
      C[(size_t)(mbase + j) * N + nn] = acc[j] + bv;
  }
}

// ---------------- small helper kernels --------------------------------------
__global__ void cvt_f32_f16(const float* __restrict__ in, _Float16* __restrict__ out, int n) {
  int i = blockIdx.x * blockDim.x + threadIdx.x;
  if (i < n) out[i] = (_Float16)in[i];
}

__global__ void gather_hidden(const int* __restrict__ items, const float* __restrict__ emb,
                              float* __restrict__ hid, _Float16* __restrict__ hid16) {
  int bn = blockIdx.x;                  // B*N blocks
  int h  = threadIdx.x;                 // 128
  int node = items[bn];
  float v = emb[(size_t)node * H_ + h];
  hid[(size_t)bn * H_ + h]   = v;
  hid16[(size_t)bn * H_ + h] = (_Float16)v;
}

// h_in = A_in @ hs_in + b_iah ; h_out = A_out @ hs_out + b_oah ; write f16 concat
__global__ void a_mix(const float* __restrict__ Afull,
                      const float* __restrict__ hs_in, const float* __restrict__ hs_out,
                      const float* __restrict__ b_iah, const float* __restrict__ b_oah,
                      _Float16* __restrict__ inputs16) {
  int b = blockIdx.x / N_;
  int i = blockIdx.x % N_;
  int h = threadIdx.x;                  // 128
  __shared__ float arow_in[N_], arow_out[N_];
  const float* Arow = Afull + ((size_t)b * N_ + i) * (2 * N_);
  if (h < N_) { arow_in[h] = Arow[h]; arow_out[h] = Arow[N_ + h]; }
  __syncthreads();
  float acc_in = b_iah[h], acc_out = b_oah[h];
  const float* hi = hs_in  + (size_t)b * N_ * H_;
  const float* ho = hs_out + (size_t)b * N_ * H_;
#pragma unroll 5
  for (int j = 0; j < N_; ++j) {
    acc_in  += arow_in[j]  * hi[(size_t)j * H_ + h];
    acc_out += arow_out[j] * ho[(size_t)j * H_ + h];
  }
  _Float16* out = inputs16 + ((size_t)b * N_ + i) * (2 * H_);
  out[h]       = (_Float16)acc_in;
  out[H_ + h]  = (_Float16)acc_out;
}

__global__ void gru_cell(const float* __restrict__ gi, const float* __restrict__ gh,
                         const float* __restrict__ hid,
                         float* __restrict__ nh, _Float16* __restrict__ nh16) {
  int bn = blockIdx.x;                  // B*N
  int h  = threadIdx.x;                 // 128
  const float* gir = gi + (size_t)bn * 3 * H_;
  const float* ghr = gh + (size_t)bn * 3 * H_;
  float i_r = gir[h], i_i = gir[H_ + h], i_n = gir[2 * H_ + h];
  float h_r = ghr[h], h_i = ghr[H_ + h], h_n = ghr[2 * H_ + h];
  float rg = 1.0f / (1.0f + expf(-(i_r + h_r)));
  float ig = 1.0f / (1.0f + expf(-(i_i + h_i)));
  float ng = tanhf(i_n + rg * h_n);
  float hv = hid[(size_t)bn * H_ + h];
  float o  = ng + ig * (hv - ng);
  nh[(size_t)bn * H_ + h]   = o;
  nh16[(size_t)bn * H_ + h] = (_Float16)o;
}

__global__ void seq_gather(const int* __restrict__ alias, const float* __restrict__ nh,
                           float* __restrict__ seq, _Float16* __restrict__ seq16,
                           float* __restrict__ ht, _Float16* __restrict__ ht16) {
  int b = blockIdx.x;                   // B
  int h = threadIdx.x;                  // 128
  const int* al = alias + (size_t)b * L_;
  int cnt = 0;
  for (int l = 0; l < L_; ++l) cnt += (al[l] > 0) ? 1 : 0;
  int last = (cnt > 0) ? (cnt - 1) : 0;
  for (int l = 0; l < L_; ++l) {
    int idx = al[l];
    float v = nh[((size_t)b * N_ + idx) * H_ + h];
    seq[((size_t)b * L_ + l) * H_ + h]   = v;
    seq16[((size_t)b * L_ + l) * H_ + h] = (_Float16)v;
  }
  int lidx = al[last];
  float hv = nh[((size_t)b * N_ + lidx) * H_ + h];
  ht[(size_t)b * H_ + h]   = hv;
  ht16[(size_t)b * H_ + h] = (_Float16)hv;
}

// alpha = sigmoid(q1+q2) @ W3^T ; a = sum_l alpha*seq*mask ; acat16 = [a | ht] f16
__global__ void attn_readout(const float* __restrict__ q1, const float* __restrict__ q2,
                             const float* __restrict__ W3, const int* __restrict__ alias,
                             const float* __restrict__ seq, const float* __restrict__ ht,
                             _Float16* __restrict__ acat16) {
  int b = blockIdx.x;                   // B
  int h = threadIdx.x;                  // 128
  __shared__ float red[H_];
  __shared__ float alpha[L_];
  float q1v = q1[(size_t)b * H_ + h];
  float w3v = W3[h];
  for (int l = 0; l < L_; ++l) {
    float q2v = q2[((size_t)b * L_ + l) * H_ + h];
    float s = 1.0f / (1.0f + expf(-(q1v + q2v)));
    red[h] = w3v * s;
    __syncthreads();
    for (int off = H_ / 2; off > 0; off >>= 1) {
      if (h < off) red[h] += red[h + off];
      __syncthreads();
    }
    if (h == 0) alpha[l] = red[0];
    __syncthreads();
  }
  const int* al = alias + (size_t)b * L_;
  float acc = 0.0f;
  for (int l = 0; l < L_; ++l) {
    if (al[l] > 0) acc += alpha[l] * seq[((size_t)b * L_ + l) * H_ + h];
  }
  _Float16* out = acat16 + (size_t)b * (2 * H_);
  out[h]      = (_Float16)acc;
  out[H_ + h] = (_Float16)ht[(size_t)b * H_ + h];
}

// ---------------- host side -------------------------------------------------
extern "C" void kernel_launch(void* const* d_in, const int* in_sizes, int n_in,
                              void* d_out, int out_size, void* d_ws, size_t ws_size,
                              hipStream_t stream) {
  (void)in_sizes; (void)n_in; (void)out_size; (void)ws_size;

  const int*   items  = (const int*)  d_in[0];
  const float* Afull  = (const float*)d_in[1];
  const int*   alias  = (const int*)  d_in[2];
  const float* emb    = (const float*)d_in[3];
  const float* w_ih   = (const float*)d_in[4];
  const float* w_hh   = (const float*)d_in[5];
  const float* b_ih   = (const float*)d_in[6];
  const float* b_hh   = (const float*)d_in[7];
  const float* b_iah  = (const float*)d_in[8];
  const float* b_oah  = (const float*)d_in[9];
  const float* W_ein  = (const float*)d_in[10];
  const float* b_ein  = (const float*)d_in[11];
  const float* W_eout = (const float*)d_in[12];
  const float* b_eout = (const float*)d_in[13];
  const float* W1     = (const float*)d_in[14];
  const float* b1     = (const float*)d_in[15];
  const float* W2     = (const float*)d_in[16];
  const float* b2     = (const float*)d_in[17];
  const float* W3     = (const float*)d_in[18];
  const float* Wt     = (const float*)d_in[19];
  const float* bt     = (const float*)d_in[20];
  float* scores = (float*)d_out;

  // ---- carve workspace ----
  char* p = (char*)d_ws;
  auto alloc = [&](size_t bytes) -> char* {
    char* r = p;
    p += (bytes + 255) & ~(size_t)255;
    return r;
  };
  _Float16* emb16    = (_Float16*)alloc((size_t)NN_ * H_ * 2);
  _Float16* wih16    = (_Float16*)alloc((size_t)3 * H_ * 2 * H_ * 2);
  _Float16* whh16    = (_Float16*)alloc((size_t)3 * H_ * H_ * 2);
  _Float16* wein16   = (_Float16*)alloc((size_t)H_ * H_ * 2);
  _Float16* weout16  = (_Float16*)alloc((size_t)H_ * H_ * 2);
  _Float16* w1_16    = (_Float16*)alloc((size_t)H_ * H_ * 2);
  _Float16* w2_16    = (_Float16*)alloc((size_t)H_ * H_ * 2);
  _Float16* wt16     = (_Float16*)alloc((size_t)H_ * 2 * H_ * 2);
  float*    hid      = (float*)   alloc((size_t)B_ * N_ * H_ * 4);
  _Float16* hid16    = (_Float16*)alloc((size_t)B_ * N_ * H_ * 2);
  float*    hs_in    = (float*)   alloc((size_t)B_ * N_ * H_ * 4);
  float*    hs_out   = (float*)   alloc((size_t)B_ * N_ * H_ * 4);
  _Float16* inputs16 = (_Float16*)alloc((size_t)B_ * N_ * 2 * H_ * 2);
  float*    gi       = (float*)   alloc((size_t)B_ * N_ * 3 * H_ * 4);
  float*    gh       = (float*)   alloc((size_t)B_ * N_ * 3 * H_ * 4);
  float*    nh       = (float*)   alloc((size_t)B_ * N_ * H_ * 4);
  _Float16* nh16     = (_Float16*)alloc((size_t)B_ * N_ * H_ * 2);
  float*    seq      = (float*)   alloc((size_t)B_ * L_ * H_ * 4);
  _Float16* seq16    = (_Float16*)alloc((size_t)B_ * L_ * H_ * 2);
  float*    ht       = (float*)   alloc((size_t)B_ * H_ * 4);
  _Float16* ht16     = (_Float16*)alloc((size_t)B_ * H_ * 2);
  float*    q1       = (float*)   alloc((size_t)B_ * H_ * 4);
  float*    q2       = (float*)   alloc((size_t)B_ * L_ * H_ * 4);
  _Float16* acat16   = (_Float16*)alloc((size_t)B_ * 2 * H_ * 2);
  float*    at       = (float*)   alloc((size_t)B_ * H_ * 4);
  _Float16* at16     = (_Float16*)alloc((size_t)B_ * H_ * 2);

  auto cvt = [&](const float* src, _Float16* dst, int n) {
    cvt_f32_f16<<<(n + 255) / 256, 256, 0, stream>>>(src, dst, n);
  };
  auto gemm = [&](const _Float16* A, const _Float16* Bt, const float* bias,
                  float* C, int M, int N, int K) {
    int tilesM = (M + 15) / 16, tilesN = (N + 15) / 16;
    int total = tilesM * tilesN;
    dim3 blk(32, 8);                             // 8 wave32 per workgroup
    int grid = (total + 7) / 8;
    if (K == 128) {
      wmma_gemm_bias_t<128><<<grid, blk, 0, stream>>>(A, Bt, bias, C, M, N,
                                                      tilesN, total);
    } else {  // K == 256
      wmma_gemm_bias_t<256><<<grid, blk, 0, stream>>>(A, Bt, bias, C, M, N,
                                                      tilesN, total);
    }
  };

  // weights + embedding -> f16
  cvt(emb,    emb16,   NN_ * H_);
  cvt(w_ih,   wih16,   3 * H_ * 2 * H_);
  cvt(w_hh,   whh16,   3 * H_ * H_);
  cvt(W_ein,  wein16,  H_ * H_);
  cvt(W_eout, weout16, H_ * H_);
  cvt(W1,     w1_16,   H_ * H_);
  cvt(W2,     w2_16,   H_ * H_);
  cvt(Wt,     wt16,    H_ * 2 * H_);

  // hidden = embedding[items]
  gather_hidden<<<B_ * N_, H_, 0, stream>>>(items, emb, hid, hid16);

  // hs_in = hidden @ W_ein^T + b_ein ; hs_out = hidden @ W_eout^T + b_eout
  gemm(hid16, wein16,  b_ein,  hs_in,  B_ * N_, H_, H_);
  gemm(hid16, weout16, b_eout, hs_out, B_ * N_, H_, H_);

  // inputs = [A_in@hs_in + b_iah | A_out@hs_out + b_oah]  (f16)
  a_mix<<<B_ * N_, H_, 0, stream>>>(Afull, hs_in, hs_out, b_iah, b_oah, inputs16);

  // gi = inputs @ w_ih^T + b_ih ; gh = hidden @ w_hh^T + b_hh
  gemm(inputs16, wih16, b_ih, gi, B_ * N_, 3 * H_, 2 * H_);
  gemm(hid16,    whh16, b_hh, gh, B_ * N_, 3 * H_, H_);

  // GRU gate math -> new hidden
  gru_cell<<<B_ * N_, H_, 0, stream>>>(gi, gh, hid, nh, nh16);

  // seq_hidden gather + ht
  seq_gather<<<B_, H_, 0, stream>>>(alias, nh, seq, seq16, ht, ht16);

  // q1 = ht @ W1^T + b1 ; q2 = seq @ W2^T + b2
  gemm(ht16,  w1_16, b1, q1, B_,      H_, H_);
  gemm(seq16, w2_16, b2, q2, B_ * L_, H_, H_);

  // attention readout -> acat16 = [a | ht]
  attn_readout<<<B_, H_, 0, stream>>>(q1, q2, W3, alias, seq, ht, acat16);

  // at = acat @ Wt^T + bt ; then f16
  gemm(acat16, wt16, bt, at, B_, H_, 2 * H_);
  cvt(at, at16, B_ * H_);

  // scores = at @ emb[1:]^T  (Bt = embedding rows 1..NN-1, contiguous f16)
  gemm(at16, emb16 + H_, nullptr, scores, B_, NN_ - 1, H_);
}
